// VQ_VAE_78589311582884
// MI455X (gfx1250) — compile-verified, matrix operand order checked by reading, not run on hardware
//
#include <hip/hip_runtime.h>
#include <hip/hip_bf16.h>

typedef __attribute__((ext_vector_type(16))) _Float16 v16h;
typedef __attribute__((ext_vector_type(8)))  _Float16 h8;
typedef __attribute__((ext_vector_type(8)))  float    v8f;
typedef __attribute__((ext_vector_type(4)))  int      v4i;

typedef __attribute__((address_space(1))) v4i* as1_v4i;
typedef __attribute__((address_space(3))) v4i* as3_v4i;

#define NB 8192          // batch
#define OBSD 64
#define CODED 128
#define NCODE 64
#define HID 256

#if __has_builtin(__builtin_amdgcn_global_load_async_to_lds_b128)
#define HAVE_ASYNC_LDS 1
#else
#define HAVE_ASYNC_LDS 0
#endif

// ---- async global -> LDS (CDNA5 GLOBAL_LOAD_ASYNC_TO_LDS_B128, ASYNCcnt) ----
__device__ __forceinline__ void async_copy16(const _Float16* g, _Float16* l) {
#if HAVE_ASYNC_LDS
  __builtin_amdgcn_global_load_async_to_lds_b128((as1_v4i)g, (as3_v4i)l, 0, 0);
#else
  *(h8*)l = *(const h8*)g;
#endif
}

__device__ __forceinline__ void wait_async() {
#if __has_builtin(__builtin_amdgcn_s_wait_asynccnt)
  __builtin_amdgcn_s_wait_asynccnt(0);
#elif HAVE_ASYNC_LDS
  asm volatile("s_wait_asynccnt 0" ::: "memory");
#endif
}

// ---- WMMA helpers (CDNA5 16x16x32 f16 -> f32) -------------------------------
__device__ __forceinline__ v16h mk16(h8 lo, h8 hi) {
  v16h r;
#pragma unroll
  for (int e = 0; e < 8; ++e) { r[e] = lo[e]; r[e + 8] = hi[e]; }
  return r;
}

// A/B fragment: lane L supplies element e at K = kk + 16*(e/8) + 8*(L/16) + e%8,
// i.e. two contiguous 8-half (16B) runs from the (row-major-in-K) operand row.
__device__ __forceinline__ v16h load_frag(const _Float16* row, int kk, int laneHi) {
  h8 lo = *(const h8*)(row + kk + 8 * laneHi);
  h8 hi = *(const h8*)(row + kk + 16 + 8 * laneHi);
  return mk16(lo, hi);
}

__device__ __forceinline__ v8f wmma_f16(v16h a, v16h b, v8f c) {
  return __builtin_amdgcn_wmma_f32_16x16x32_f16(false, a, false, b, (short)0, c,
                                                false, false);
}

// ---- prep kernels: build transposed f16 weight copies in workspace ----------
__global__ __launch_bounds__(256) void prep_enc_kernel(
    const float* __restrict__ enc_w1, const float* __restrict__ enc_b1,
    const float* __restrict__ enc_w2, float* __restrict__ W0,
    float* __restrict__ R, _Float16* __restrict__ w2t) {
  int g = blockIdx.x * 256 + threadIdx.x;
  if (g < HID) W0[g] = enc_w1[g];                       // enc_w1 row 0
  if (g < OBSD * HID) {                                 // R[p][j] = w1[1+p][j]+b1[j]
    int p = g >> 8, j = g & 255;
    R[g] = enc_w1[(1 + p) * HID + j] + enc_b1[j];
  }
  if (g < CODED * HID) {                                // w2t[d][j] = enc_w2[j][d]
    int d = g >> 8, j = g & 255;
    w2t[g] = (_Float16)enc_w2[j * CODED + d];
  }
}

__global__ __launch_bounds__(256) void prep_emb_kernel(
    const float* __restrict__ emb_w, _Float16* __restrict__ wptH,
    float* __restrict__ wptF, float* __restrict__ w2s) {
  int g = blockIdx.x * 256 + threadIdx.x;
  if (g < OBSD * NCODE * CODED) {                       // wpt[p][k][d] = w[d,p,k]
    int d = g & 127, k = (g >> 7) & 63, p = g >> 13;
    float v = emb_w[(size_t)d * (OBSD * NCODE) + p * NCODE + k];
    wptH[g] = (_Float16)v;
    wptF[g] = v;
  }
  if (g < OBSD * NCODE) {                               // w2[p][k] = sum_d w^2
    int p = g >> 6, k = g & 63;
    float s = 0.f;
    for (int d = 0; d < CODED; ++d) {
      float v = emb_w[(size_t)d * (OBSD * NCODE) + p * NCODE + k];
      s = fmaf(v, v, s);
    }
    w2s[g] = s;
  }
}

__global__ __launch_bounds__(256) void prep_dec_kernel(
    const float* __restrict__ dec_w1, const float* __restrict__ dec_w2,
    _Float16* __restrict__ dw1t, _Float16* __restrict__ dw2t) {
  int g = blockIdx.x * 256 + threadIdx.x;
  if (g < HID * 8192) {                                 // dw1t[j][kk] = dec_w1[kk][j]
    int j = g >> 13, kk = g & 8191;
    dw1t[g] = (_Float16)dec_w1[(size_t)kk * HID + j];
  }
  if (g < OBSD * HID) {                                 // dw2t[n][j] = dec_w2[j][n]
    int n = g >> 8, j = g & 255;
    dw2t[g] = (_Float16)dec_w2[j * OBSD + n];
  }
}

// ---- fused encoder + vector-quantize ---------------------------------------
// grid: (64 p, 512 b-tiles), 256 threads = 8 wave32.
__global__ __launch_bounds__(256) void encode_vq_kernel(
    const float* __restrict__ obs, const float* __restrict__ W0,
    const float* __restrict__ R, const _Float16* __restrict__ w2t,
    const float* __restrict__ enc_b2, const _Float16* __restrict__ wptH,
    const float* __restrict__ wptF, const float* __restrict__ w2s,
    float* __restrict__ ze, float* __restrict__ emb, int* __restrict__ idxb) {
  __shared__ _Float16 Bl[128][264];  // enc_w2t panel (async-staged), padded rows
  __shared__ _Float16 Wl[64][136];   // codebook panel for this p, padded rows
  __shared__ _Float16 Ah[16][264];   // h tile
  __shared__ _Float16 Zh[16][136];   // z tile f16
  __shared__ float    Zf[16][129];   // z tile f32
  __shared__ float    Dist[16][68];
  __shared__ float    z2s[16];
  __shared__ int      idxs[16];

  const int p = blockIdx.x;
  const int b0 = blockIdx.y * 16;
  const int t = threadIdx.x;
  const int wave = t >> 5, lane = t & 31;
  const int laneHi = lane >> 4, lane16 = lane & 15;

  // 0. kick off async DMA of both B panels into LDS (overlaps h computation)
  {
    const _Float16* wp = wptH + (size_t)p * NCODE * CODED;
#pragma unroll
    for (int i = 0; i < 4; ++i) {            // 64 rows * 16 segs of 8 halves
      int idx = t + i * 256;
      int r = idx >> 4, seg = idx & 15;
      async_copy16(wp + r * CODED + seg * 8, &Wl[r][seg * 8]);
    }
#pragma unroll
    for (int i = 0; i < 16; ++i) {           // 128 rows * 32 segs of 8 halves
      int idx = t + i * 256;
      int r = idx >> 5, seg = idx & 31;
      async_copy16(w2t + r * HID + seg * 8, &Bl[r][seg * 8]);
    }
  }

  // 1. h[b,p,j] = relu(obs*W0 + R)  (one-hot structure collapses the GEMM)
  {
    int row = t >> 4;
    int j0 = (t & 15) * 16;
    float ob = obs[(size_t)(b0 + row) * OBSD + p];
    __builtin_prefetch(w2s + p * NCODE, 0, 1);
#pragma unroll
    for (int jj = 0; jj < 16; ++jj) {
      int j = j0 + jj;
      float hv = fmaf(ob, W0[j], R[p * HID + j]);
      Ah[row][j] = (_Float16)(hv > 0.f ? hv : 0.f);
    }
  }
  wait_async();
  __syncthreads();

  // 2. z = h @ enc_w2 + b2  (WMMA, K=256; wave owns 16 d-columns; all-LDS feed)
  {
    v8f acc = {};
    const _Float16* arow = &Ah[lane16][0];
    const _Float16* brow = &Bl[wave * 16 + lane16][0];
#pragma unroll
    for (int kk = 0; kk < HID; kk += 32)
      acc = wmma_f16(load_frag(arow, kk, laneHi), load_frag(brow, kk, laneHi), acc);
    int d = wave * 16 + lane16;
    float bias = enc_b2[d];
#pragma unroll
    for (int v = 0; v < 8; ++v) {
      int br = v + 8 * laneHi;
      float zv = acc[v] + bias;
      Zf[br][d] = zv;
      Zh[br][d] = (_Float16)zv;
      ze[(size_t)(b0 + br) * (CODED * OBSD) + (size_t)d * OBSD + p] = zv;
    }
  }
  __syncthreads();

  // 3. z2[b] = ||z||^2
  if (t < 16) {
    float s = 0.f;
#pragma unroll 4
    for (int d = 0; d < CODED; ++d) { float z = Zf[t][d]; s = fmaf(z, z, s); }
    z2s[t] = s;
  }
  __syncthreads();

  // 4. cross = z @ w[:,p,:]  (WMMA, K=128; waves 0..3 own the 64 codes)
  if (wave < 4) {
    v8f cc = {};
    const _Float16* arow = &Zh[lane16][0];
    const _Float16* brow = &Wl[wave * 16 + lane16][0];
#pragma unroll
    for (int kk = 0; kk < CODED; kk += 32)
      cc = wmma_f16(load_frag(arow, kk, laneHi), load_frag(brow, kk, laneHi), cc);
    int k = wave * 16 + lane16;
    float w2v = w2s[p * NCODE + k];
#pragma unroll
    for (int v = 0; v < 8; ++v) {
      int br = v + 8 * laneHi;
      Dist[br][k] = z2s[br] + w2v - 2.f * cc[v];
    }
  }
  __syncthreads();

  // 5. argmin over 64 codes (first-min tie-break, matches jnp.argmin)
  if (t < 16) {
    float best = Dist[t][0];
    int bi = 0;
    for (int k = 1; k < NCODE; ++k) {
      float dv = Dist[t][k];
      if (dv < best) { best = dv; bi = k; }
    }
    idxs[t] = bi;
    idxb[(size_t)(b0 + t) * OBSD + p] = bi;
  }
  __syncthreads();

  // 6. emb[b,d,p] = w[d,p,idx]  (exact f32 gather)
  {
    int row = t >> 4;
    int d0 = (t & 15) * 8;
    int ic = idxs[row];
    const float* src = wptF + ((size_t)p * NCODE + ic) * CODED + d0;
    size_t obase = (size_t)(b0 + row) * (CODED * OBSD) + p;
#pragma unroll
    for (int dd = 0; dd < 8; ++dd)
      emb[obase + (size_t)(d0 + dd) * OBSD] = src[dd];
  }
}

// ---- Mtab[p][k][j] = sum_d w[d,p,k] * dec_w1[p*128+d][j]  (WMMA) ------------
__global__ __launch_bounds__(256) void mtab_kernel(
    const _Float16* __restrict__ wptH, const _Float16* __restrict__ dw1t,
    float* __restrict__ Mtab) {
  int p = blockIdx.x;
  int t = threadIdx.x;
  int wave = t >> 5, lane = t & 31, laneHi = lane >> 4, lane16 = lane & 15;
#pragma unroll 1
  for (int i = 0; i < 8; ++i) {
    int id = wave * 8 + i;
    int tm = id >> 4;   // k-tile 0..3
    int tn = id & 15;   // j-tile 0..15
    const _Float16* arow = wptH + ((size_t)p * NCODE + tm * 16 + lane16) * CODED;
    const _Float16* brow = dw1t + (size_t)(tn * 16 + lane16) * 8192 + p * CODED;
    v8f acc = {};
#pragma unroll
    for (int kk = 0; kk < CODED; kk += 32)
      acc = wmma_f16(load_frag(arow, kk, laneHi), load_frag(brow, kk, laneHi), acc);
#pragma unroll
    for (int v = 0; v < 8; ++v) {
      int kr = tm * 16 + v + 8 * laneHi;
      Mtab[((size_t)p * NCODE + kr) * HID + tn * 16 + lane16] = acc[v];
    }
  }
}

// ---- h2[b,j] = relu(b1 + sum_p Mtab[p, idx[b,p], j])  (L2-resident gather) --
__global__ __launch_bounds__(256) void h2_kernel(
    const float* __restrict__ Mtab, const int* __restrict__ idxb,
    const float* __restrict__ dec_b1, _Float16* __restrict__ h2h) {
  int b = blockIdx.x;
  int j = threadIdx.x;
  float acc = dec_b1[j];
  const int* ib = idxb + (size_t)b * OBSD;
#pragma unroll 4
  for (int p = 0; p < OBSD; ++p) {
    int ic = ib[p];                      // scalar per block
    acc += Mtab[((size_t)(p * NCODE + ic)) * HID + j];
  }
  acc = acc > 0.f ? acc : 0.f;
  h2h[(size_t)b * HID + j] = (_Float16)acc;
}

// ---- recon = h2 @ dec_w2 + b2  (WMMA) ---------------------------------------
__global__ __launch_bounds__(256) void recon_kernel(
    const _Float16* __restrict__ h2h, const _Float16* __restrict__ dw2t,
    const float* __restrict__ dec_b2, float* __restrict__ recon) {
  int b0 = blockIdx.x * 32;
  int t = threadIdx.x;
  int wave = t >> 5, lane = t & 31, laneHi = lane >> 4, lane16 = lane & 15;
  int tm = wave >> 2, tn = wave & 3;
  const _Float16* arow = h2h + (size_t)(b0 + tm * 16 + lane16) * HID;
  const _Float16* brow = dw2t + (size_t)(tn * 16 + lane16) * HID;
  __builtin_prefetch(arow, 0, 0);
  v8f acc = {};
#pragma unroll
  for (int kk = 0; kk < HID; kk += 32)
    acc = wmma_f16(load_frag(arow, kk, laneHi), load_frag(brow, kk, laneHi), acc);
  int n = tn * 16 + lane16;
  float bias = dec_b2[n];
#pragma unroll
  for (int v = 0; v < 8; ++v) {
    int b = b0 + tm * 16 + v + 8 * laneHi;
    recon[(size_t)b * OBSD + n] = acc[v] + bias;
  }
}

extern "C" void kernel_launch(void* const* d_in, const int* in_sizes, int n_in,
                              void* d_out, int out_size, void* d_ws, size_t ws_size,
                              hipStream_t stream) {
  const float* obs    = (const float*)d_in[0];
  const float* enc_w1 = (const float*)d_in[1];
  const float* enc_b1 = (const float*)d_in[2];
  const float* enc_w2 = (const float*)d_in[3];
  const float* enc_b2 = (const float*)d_in[4];
  const float* emb_w  = (const float*)d_in[5];
  const float* dec_w1 = (const float*)d_in[6];
  const float* dec_b1 = (const float*)d_in[7];
  const float* dec_w2 = (const float*)d_in[8];
  const float* dec_b2 = (const float*)d_in[9];

  float* recon = (float*)d_out;                           // (8192, 64)
  float* ze    = recon + (size_t)NB * OBSD;               // (8192, 128, 64)
  float* emb   = ze + (size_t)NB * CODED * OBSD;          // (8192, 128, 64)

  char* w = (char*)d_ws;                                  // ~18 MB of scratch
  float*    W0   = (float*)(w + 0);                       // 1 KB
  float*    R    = (float*)(w + 1024);                    // 64 KB
  _Float16* w2t  = (_Float16*)(w + 66560);                // 64 KB
  _Float16* wptH = (_Float16*)(w + 132096);               // 1 MB
  float*    wptF = (float*)(w + 1180672);                 // 2 MB
  float*    w2s  = (float*)(w + 3277824);                 // 16 KB
  _Float16* dw1t = (_Float16*)(w + 3294208);              // 4 MB
  _Float16* dw2t = (_Float16*)(w + 7488512);              // 32 KB
  float*    Mtab = (float*)(w + 7521280);                 // 4 MB
  int*      idxb = (int*)(w + 11715584);                  // 2 MB
  _Float16* h2h  = (_Float16*)(w + 13812736);             // 4 MB

  prep_enc_kernel<<<128, 256, 0, stream>>>(enc_w1, enc_b1, enc_w2, W0, R, w2t);
  prep_emb_kernel<<<2048, 256, 0, stream>>>(emb_w, wptH, wptF, w2s);
  prep_dec_kernel<<<8192, 256, 0, stream>>>(dec_w1, dec_w2, dw1t, dw2t);

  encode_vq_kernel<<<dim3(64, 512), 256, 0, stream>>>(
      obs, W0, R, w2t, enc_b2, wptH, wptF, w2s, ze, emb, idxb);

  mtab_kernel<<<64, 256, 0, stream>>>(wptH, dw1t, Mtab);
  h2_kernel<<<8192, 256, 0, stream>>>(Mtab, idxb, dec_b1, h2h);
  recon_kernel<<<256, 256, 0, stream>>>(h2h, dw2t, dec_b2, recon);
}
